// GroupQueryAttention_10849087389768
// MI455X (gfx1250) — matrix operand order repaired and checked by reading, not compiled, not run on hardware
//
#include <hip/hip_runtime.h>
#include <hip/hip_bf16.h>

// GQA forward for MI455X (gfx1250): all matmuls via V_WMMA_F32_16X16X32_BF16.
// Workspace requirement: ~136 MB.

#define B_   2
#define S_   2048
#define D_   2048
#define NH_  16
#define NG_  4
#define KVH_ (NH_/NG_)   // 4
#define HD_  (D_/NH_)    // 128
#define DK_  (D_/NG_)    // 512

typedef __bf16 bf16_t;
typedef __attribute__((ext_vector_type(16))) __bf16 v16bf;
typedef __attribute__((ext_vector_type(8)))  __bf16 v8bf;
typedef __attribute__((ext_vector_type(8)))  float  v8f;

static __device__ __forceinline__ v16bf cat8(v8bf lo, v8bf hi) {
    return __builtin_shufflevector(lo, hi, 0,1,2,3,4,5,6,7,8,9,10,11,12,13,14,15);
}
static __device__ __forceinline__ v8f wmma_bf16(v16bf a, v16bf b, v8f c) {
    // 8 args: (neg_a, A, neg_b, B, c_mod, C, reuse_a, reuse_b)
    return __builtin_amdgcn_wmma_f32_16x16x32_bf16(false, a, false, b, (short)0, c, false, false);
}

// ---------------------------------------------------------------- converters
__global__ void cvt_f32_bf16(const float* __restrict__ src, bf16_t* __restrict__ dst, size_t n) {
    size_t i = (size_t)blockIdx.x * blockDim.x + threadIdx.x;
    if (i < n) dst[i] = (bf16_t)src[i];
}

// ------------------------------------------------------- generic WMMA GEMM
// C[M,N] (f32) = A[M,K] (bf16 row-major) * Bw[N,K]^T (bf16 row-major).
// One wave (32 threads) computes a 64x64 tile as 4x4 grid of 16x16x32 WMMAs.
__global__ __launch_bounds__(32)
void gemm_bf16_nt(const bf16_t* __restrict__ A, const bf16_t* __restrict__ Bw,
                  float* __restrict__ C, int M, int N, int K)
{
    const int lane = threadIdx.x & 31;
    const int l16  = lane & 15;
    const int half = lane >> 4;
    const int n0 = blockIdx.x * 64;
    const int m0 = blockIdx.y * 64;

    v8f acc[4][4] = {};

    for (int k0 = 0; k0 < K; k0 += 32) {
        v16bf af[4], bfr[4];
#pragma unroll
        for (int i = 0; i < 4; ++i) {
            // A fragment: lane holds row m0+16i+l16; K elems {k0+8h..+7, k0+16+8h..+7}
            const bf16_t* ap = A + (size_t)(m0 + 16*i + l16) * K + k0 + 8*half;
            v8bf lo = *reinterpret_cast<const v8bf*>(ap);
            v8bf hi = *reinterpret_cast<const v8bf*>(ap + 16);
            af[i] = cat8(lo, hi);
            // B fragment: lane holds col n0+16i+l16; K elems {k0+8h..+7, k0+16+8h..+7}
            const bf16_t* bp = Bw + (size_t)(n0 + 16*i + l16) * K + k0 + 8*half;
            bfr[i] = cat8(*reinterpret_cast<const v8bf*>(bp), *reinterpret_cast<const v8bf*>(bp + 16));
        }
#pragma unroll
        for (int i = 0; i < 4; ++i)
#pragma unroll
            for (int j = 0; j < 4; ++j)
                acc[i][j] = wmma_bf16(af[i], bfr[j], acc[i][j]);
    }
    // C layout: lane L, VGPR r -> row r+8*(L/16), col L%16
#pragma unroll
    for (int i = 0; i < 4; ++i)
#pragma unroll
        for (int j = 0; j < 4; ++j)
#pragma unroll
            for (int r = 0; r < 8; ++r)
                C[(size_t)(m0 + 16*i + r + 8*half) * N + (n0 + 16*j + l16)] = acc[i][j][r];
}

// --------------------------------------------------------- RoPE + rearrange
// Reference repeats K to full D *before* RoPE, so frequency index is the
// post-repeat absolute dim (h*HD + hd): roped K differs per head.
__global__ void rope_q_kernel(const float* __restrict__ Qf, bf16_t* __restrict__ Qh) {
    size_t idx = (size_t)blockIdx.x * blockDim.x + threadIdx.x;
    const size_t total = (size_t)B_ * NH_ * S_ * (HD_/2);
    if (idx >= total) return;
    int ip = (int)(idx % (HD_/2)); size_t t = idx / (HD_/2);
    int s  = (int)(t % S_); t /= S_;
    int h  = (int)(t % NH_); int b = (int)(t / NH_);
    int d  = h * HD_ + 2*ip;
    const float* src = Qf + ((size_t)b*S_ + s)*D_ + d;
    float xr = src[0], xi = src[1];
    float fr = __powf(10000.0f, -(float)d / (float)D_);
    float c, sn; __sincosf((float)s * fr, &sn, &c);
    bf16_t* o = Qh + (((size_t)b*NH_ + h)*S_ + s)*HD_ + 2*ip;
    o[0] = (bf16_t)(xr*c - xi*sn);
    o[1] = (bf16_t)(xr*sn + xi*c);
}

__global__ void rope_k_kernel(const float* __restrict__ Kf, bf16_t* __restrict__ Kh) {
    size_t idx = (size_t)blockIdx.x * blockDim.x + threadIdx.x;
    const size_t total = (size_t)B_ * NH_ * S_ * (HD_/2);
    if (idx >= total) return;
    int ip = (int)(idx % (HD_/2)); size_t t = idx / (HD_/2);
    int s  = (int)(t % S_); t /= S_;
    int h  = (int)(t % NH_); int b = (int)(t / NH_);
    int d  = h * HD_ + 2*ip;                 // post-repeat dim -> frequency
    int kvh = h / NG_;
    const float* src = Kf + ((size_t)b*S_ + s)*DK_ + kvh*HD_ + 2*ip;
    float xr = src[0], xi = src[1];
    float fr = __powf(10000.0f, -(float)d / (float)D_);
    float c, sn; __sincosf((float)s * fr, &sn, &c);
    bf16_t* o = Kh + (((size_t)b*NH_ + h)*S_ + s)*HD_ + 2*ip;
    o[0] = (bf16_t)(xr*c - xi*sn);
    o[1] = (bf16_t)(xr*sn + xi*c);
}

// V^T per group: Vt[b,kvh,hd,s] (bf16) so PV B-fragments are contiguous loads.
__global__ void v_trans_kernel(const float* __restrict__ Vf, bf16_t* __restrict__ Vt) {
    size_t idx = (size_t)blockIdx.x * blockDim.x + threadIdx.x;
    const size_t total = (size_t)B_ * KVH_ * HD_ * S_;
    if (idx >= total) return;
    int s  = (int)(idx % S_); size_t t = idx / S_;
    int hd = (int)(t % HD_); t /= HD_;
    int kvh = (int)(t % KVH_); int b = (int)(t / KVH_);
    Vt[idx] = (bf16_t)Vf[((size_t)b*S_ + s)*DK_ + kvh*HD_ + hd];
}

// ----------------------------------------------------- flash attention core
// One wave per (b, h, 16-query tile). Causal loop over 32-key tiles.
__global__ __launch_bounds__(32)
void attn_kernel(const bf16_t* __restrict__ Qh, const bf16_t* __restrict__ Kh,
                 const bf16_t* __restrict__ Vt, bf16_t* __restrict__ attnB)
{
    __shared__ bf16_t plds[16*32];           // P tile staging (C-layout -> A-layout)
    const int lane = threadIdx.x & 31;
    const int l16  = lane & 15;
    const int half = lane >> 4;
    const int q0 = blockIdx.x * 16;
    const int h  = blockIdx.y;
    const int b  = blockIdx.z;
    const int kvh = h / NG_;

    const bf16_t* Qp = Qh + ((size_t)b*NH_ + h) * (size_t)S_ * HD_;
    const bf16_t* Kp = Kh + ((size_t)b*NH_ + h) * (size_t)S_ * HD_;
    const bf16_t* Vp = Vt + ((size_t)b*KVH_ + kvh) * (size_t)HD_ * S_;

    // Persistent Q fragments: 16 rows x 128 (4 K-chunks of 32)
    v16bf qf[4];
#pragma unroll
    for (int kc = 0; kc < 4; ++kc) {
        const bf16_t* qp = Qp + (size_t)(q0 + l16) * HD_ + kc*32 + 8*half;
        v8bf lo = *reinterpret_cast<const v8bf*>(qp);
        v8bf hi = *reinterpret_cast<const v8bf*>(qp + 16);
        qf[kc] = cat8(lo, hi);
    }

    v8f oacc[8] = {};                        // 16 x 128 f32 output accumulator
    float mrow[8], lrow[8];
#pragma unroll
    for (int r = 0; r < 8; ++r) { mrow[r] = -3.0e38f; lrow[r] = 0.0f; }
    const float sscale = rsqrtf((float)D_);  // reference divides by sqrt(D)
    const int qmax = q0 + 15;

    for (int kb = 0; kb <= qmax; kb += 32) {
        // ---- scores: two 16x16 tiles, K-dim 128 => 8 WMMAs
        v8f sc0 = {}, sc1 = {};
        {
            const bf16_t* kp0 = Kp + (size_t)(kb + l16) * HD_;
            const bf16_t* kp1 = Kp + (size_t)(kb + 16 + l16) * HD_;
#pragma unroll
            for (int kc = 0; kc < 4; ++kc) {
                v16bf kf0 = cat8(*reinterpret_cast<const v8bf*>(kp0 + kc*32 + 8*half), *reinterpret_cast<const v8bf*>(kp0 + kc*32 + 8*half + 16));
                sc0 = wmma_bf16(qf[kc], kf0, sc0);
                v16bf kf1 = cat8(*reinterpret_cast<const v8bf*>(kp1 + kc*32 + 8*half), *reinterpret_cast<const v8bf*>(kp1 + kc*32 + 8*half + 16));
                sc1 = wmma_bf16(qf[kc], kf1, sc1);
            }
        }
        // ---- scale + causal mask + online softmax
        float p0[8], p1[8], rmax[8], rsum[8], scal[8];
#pragma unroll
        for (int r = 0; r < 8; ++r) {
            int q = q0 + r + 8*half;
            float s0 = sc0[r] * sscale; if (kb + l16 > q)      s0 = -3.0e38f;
            float s1 = sc1[r] * sscale; if (kb + 16 + l16 > q) s1 = -3.0e38f;
            p0[r] = s0; p1[r] = s1; rmax[r] = fmaxf(s0, s1);
        }
#pragma unroll
        for (int o = 1; o < 16; o <<= 1)
#pragma unroll
            for (int r = 0; r < 8; ++r)
                rmax[r] = fmaxf(rmax[r], __shfl_xor(rmax[r], o, 16));
#pragma unroll
        for (int r = 0; r < 8; ++r) {
            float mnew = fmaxf(mrow[r], rmax[r]);
            scal[r] = __expf(mrow[r] - mnew);
            mrow[r] = mnew;
            p0[r] = __expf(p0[r] - mnew);
            p1[r] = __expf(p1[r] - mnew);
            rsum[r] = p0[r] + p1[r];
        }
#pragma unroll
        for (int o = 1; o < 16; o <<= 1)
#pragma unroll
            for (int r = 0; r < 8; ++r)
                rsum[r] += __shfl_xor(rsum[r], o, 16);
        v8f sv;
#pragma unroll
        for (int r = 0; r < 8; ++r) { lrow[r] = lrow[r]*scal[r] + rsum[r]; sv[r] = scal[r]; }
#pragma unroll
        for (int ct = 0; ct < 8; ++ct) oacc[ct] *= sv;

        // ---- P: C-layout -> bf16 A-layout via LDS transpose
        __syncthreads();
#pragma unroll
        for (int r = 0; r < 8; ++r) {
            plds[(r + 8*half)*32 + l16]      = (bf16_t)p0[r];
            plds[(r + 8*half)*32 + 16 + l16] = (bf16_t)p1[r];
        }
        __syncthreads();
        v16bf pf;
        {
            const bf16_t* pp = &plds[l16*32 + 8*half];
            v8bf lo = *reinterpret_cast<const v8bf*>(pp);
            v8bf hi = *reinterpret_cast<const v8bf*>(pp + 16);
            pf = cat8(lo, hi);
        }
        // ---- PV: 8 WMMAs over V^T (K-dim contiguous along sequence)
#pragma unroll
        for (int ct = 0; ct < 8; ++ct) {
            const bf16_t* vp = Vp + (size_t)(ct*16 + l16) * S_ + kb + 8*half;
            v16bf vf = cat8(*reinterpret_cast<const v8bf*>(vp), *reinterpret_cast<const v8bf*>(vp + 16));
            oacc[ct] = wmma_bf16(pf, vf, oacc[ct]);
        }
    }

    // ---- normalize and scatter to [B,S,D] bf16 (head-concat layout)
#pragma unroll
    for (int ct = 0; ct < 8; ++ct)
#pragma unroll
        for (int r = 0; r < 8; ++r) {
            int q = q0 + r + 8*half;
            attnB[((size_t)b*S_ + q)*D_ + h*HD_ + ct*16 + l16] =
                (bf16_t)(oacc[ct][r] / lrow[r]);
        }
}

// ------------------------------------------------------------------ launcher
extern "C" void kernel_launch(void* const* d_in, const int* in_sizes, int n_in,
                              void* d_out, int out_size, void* d_ws, size_t ws_size,
                              hipStream_t stream)
{
    const float* x  = (const float*)d_in[0];
    const float* Wq = (const float*)d_in[1];
    const float* Wk = (const float*)d_in[2];
    const float* Wv = (const float*)d_in[3];
    const float* Wo = (const float*)d_in[4];
    // d_in[5] (causal mask) is realized analytically in attn_kernel.
    (void)in_sizes; (void)n_in; (void)out_size; (void)ws_size;

    char* ws = (char*)d_ws;
    size_t off = 0;
    auto carve = [&](size_t bytes) -> void* {
        void* p = (void*)(ws + off);
        off += (bytes + 255) & ~(size_t)255;
        return p;
    };
    const size_t M = (size_t)B_ * S_;   // 4096 rows

    bf16_t* xb    = (bf16_t*)carve(M * D_ * 2);
    bf16_t* Wqb   = (bf16_t*)carve((size_t)D_  * D_ * 2);
    bf16_t* Wkb   = (bf16_t*)carve((size_t)DK_ * D_ * 2);
    bf16_t* Wvb   = (bf16_t*)carve((size_t)DK_ * D_ * 2);
    bf16_t* Wob   = (bf16_t*)carve((size_t)D_  * D_ * 2);
    float*  Qf    = (float*) carve(M * D_  * 4);
    float*  Kf    = (float*) carve(M * DK_ * 4);
    float*  Vf    = (float*) carve(M * DK_ * 4);
    bf16_t* Qhb   = (bf16_t*)carve(M * D_  * 2);  // [B,NH,S,HD]
    bf16_t* Khb   = (bf16_t*)carve(M * D_  * 2);  // [B,NH,S,HD] (per-head roped K)
    bf16_t* Vtb   = (bf16_t*)carve(M * DK_ * 2);  // [B,KVH,HD,S]
    bf16_t* attnB = (bf16_t*)carve(M * D_  * 2);  // [B,S,D]

    auto gs = [](size_t n) { return dim3((unsigned)((n + 255) / 256)); };

    // 1) f32 -> bf16
    cvt_f32_bf16<<<gs(M * D_),              256, 0, stream>>>(x,  xb,  M * D_);
    cvt_f32_bf16<<<gs((size_t)D_  * D_),    256, 0, stream>>>(Wq, Wqb, (size_t)D_  * D_);
    cvt_f32_bf16<<<gs((size_t)DK_ * D_),    256, 0, stream>>>(Wk, Wkb, (size_t)DK_ * D_);
    cvt_f32_bf16<<<gs((size_t)DK_ * D_),    256, 0, stream>>>(Wv, Wvb, (size_t)DK_ * D_);
    cvt_f32_bf16<<<gs((size_t)D_  * D_),    256, 0, stream>>>(Wo, Wob, (size_t)D_  * D_);

    // 2) projections (C = A * W^T)
    gemm_bf16_nt<<<dim3(D_/64,  (unsigned)(M/64)), 32, 0, stream>>>(xb, Wqb, Qf, (int)M, D_,  D_);
    gemm_bf16_nt<<<dim3(DK_/64, (unsigned)(M/64)), 32, 0, stream>>>(xb, Wkb, Kf, (int)M, DK_, D_);
    gemm_bf16_nt<<<dim3(DK_/64, (unsigned)(M/64)), 32, 0, stream>>>(xb, Wvb, Vf, (int)M, DK_, D_);

    // 3) RoPE + head layouts
    rope_q_kernel <<<gs((size_t)B_*NH_ *S_*(HD_/2)), 256, 0, stream>>>(Qf, Qhb);
    rope_k_kernel <<<gs((size_t)B_*NH_ *S_*(HD_/2)), 256, 0, stream>>>(Kf, Khb);
    v_trans_kernel<<<gs((size_t)B_*KVH_*HD_*S_),     256, 0, stream>>>(Vf, Vtb);

    // 4) flash attention
    attn_kernel<<<dim3(S_/16, NH_, B_), 32, 0, stream>>>(Qhb, Khb, Vtb, attnB);

    // 5) output projection -> f32 d_out
    gemm_bf16_nt<<<dim3(D_/64, (unsigned)(M/64)), 32, 0, stream>>>(attnB, Wob, (float*)d_out,
                                                                   (int)M, D_, D_);
}
